// Llama4TextMoe_29892972380385
// MI455X (gfx1250) — matrix operand order repaired and checked by reading, not compiled, not run on hardware
//
#include <hip/hip_runtime.h>
#include <hip/hip_bf16.h>

// ---------------- problem constants (from reference) ----------------
constexpr int Bc = 2, Sc = 2048, Dc = 2048, Fc = 4096, Ec = 8;
constexpr int Tc = Bc * Sc;          // 4096 tokens
constexpr int F2c = 2 * Fc;          // 8192 (gate|up)

// ---------------- GEMM tiling ----------------
#define BM 128
#define BN 256
#define BK 32
#define LDK 40                        // padded LDS row (bf16 elems) -> 80B, 16B-aligned

typedef __attribute__((ext_vector_type(16))) __bf16 v16bf;
typedef __attribute__((ext_vector_type(8)))  __bf16 v8bf;
typedef __attribute__((ext_vector_type(8)))  float  v8f;

// =====================================================================
// Router: one block per token, top-1 of 8 experts + sigmoid gate
// =====================================================================
__global__ __launch_bounds__(256) void router_kernel(
    const float* __restrict__ x, const float* __restrict__ rw,
    int* __restrict__ eid, float* __restrict__ gate, int* __restrict__ counts)
{
    const int t = blockIdx.x;
    const int tid = threadIdx.x;
    const float* xr = x + (long long)t * Dc;
    float acc[Ec];
#pragma unroll
    for (int e = 0; e < Ec; ++e) acc[e] = 0.f;
    for (int d = tid; d < Dc; d += 256) {
        float xv = xr[d];
#pragma unroll
        for (int e = 0; e < Ec; ++e) acc[e] += xv * rw[e * Dc + d];
    }
    __shared__ float red[256];
    __shared__ float logits[Ec];
    for (int e = 0; e < Ec; ++e) {
        red[tid] = acc[e];
        __syncthreads();
        for (int s = 128; s > 0; s >>= 1) {
            if (tid < s) red[tid] += red[tid + s];
            __syncthreads();
        }
        if (tid == 0) logits[e] = red[0];
        __syncthreads();
    }
    if (tid == 0) {
        int best = 0; float bv = logits[0];
#pragma unroll
        for (int e = 1; e < Ec; ++e) if (logits[e] > bv) { bv = logits[e]; best = e; }
        eid[t]  = best;
        gate[t] = 1.f / (1.f + __expf(-bv));
        atomicAdd(&counts[best], 1);
    }
}

__global__ void scan_kernel(const int* __restrict__ counts,
                            int* __restrict__ offsets, int* __restrict__ cursor)
{
    if (threadIdx.x == 0) {
        int s = 0;
        for (int e = 0; e < Ec; ++e) { offsets[e] = s; s += counts[e]; cursor[e] = 0; }
    }
}

__global__ void scatter_kernel(const int* __restrict__ eid, const int* __restrict__ offsets,
                               int* __restrict__ cursor, int* __restrict__ tlist)
{
    int t = blockIdx.x * blockDim.x + threadIdx.x;
    if (t < Tc) {
        int e = eid[t];
        int p = atomicAdd(&cursor[e], 1);
        tlist[offsets[e] + p] = t;
    }
}

// x -> bf16 (shared path) and x*gate -> bf16 (routed path), 4 elems/thread
__global__ __launch_bounds__(256) void prep_kernel(
    const float* __restrict__ x, const float* __restrict__ gate,
    __bf16* __restrict__ xb, __bf16* __restrict__ xs)
{
    long long i = ((long long)blockIdx.x * blockDim.x + threadIdx.x) * 4;
    if (i >= (long long)Tc * Dc) return;
    float4 f = *(const float4*)(x + i);
    float g = gate[i / Dc];
    xb[i + 0] = (__bf16)f.x; xb[i + 1] = (__bf16)f.y;
    xb[i + 2] = (__bf16)f.z; xb[i + 3] = (__bf16)f.w;
    xs[i + 0] = (__bf16)(f.x * g); xs[i + 1] = (__bf16)(f.y * g);
    xs[i + 2] = (__bf16)(f.z * g); xs[i + 3] = (__bf16)(f.w * g);
}

// silu(g)*u on merged [gate|up] rows (bf16 in, bf16 out), 4 cols/thread
__global__ __launch_bounds__(256) void silu_kernel(
    const __bf16* __restrict__ Hr, __bf16* __restrict__ Hs)
{
    long long i = ((long long)blockIdx.x * blockDim.x + threadIdx.x) * 4;
    if (i >= (long long)Tc * Fc) return;
    long long p = i / Fc; int f = (int)(i % Fc);
    const __bf16* row = Hr + p * (long long)F2c;
    __bf16* outp = Hs + p * (long long)Fc + f;
#pragma unroll
    for (int j = 0; j < 4; ++j) {
        float g = (float)row[f + j];
        float u = (float)row[Fc + f + j];
        float s = g / (1.f + __expf(-g));   // silu
        outp[j] = (__bf16)(s * u);
    }
}

// =====================================================================
// Grouped bf16 GEMM: C[M x N] = A[M x K](bf16) * B[K x N](fp32->bf16)
//   CMODE 0: store bf16       (gate_up -> Hr)
//   CMODE 1: store f32  "="   (routed down -> out, scatter rows)
//   CMODE 2: store f32  "+="  (shared down -> out)
// 256 threads = 8 wave32 (2x4). Block tile 128x256, wave tile 64x64
// (4x4 v_wmma_f32_16x16x32_bf16 per k-step). Double-buffered LDS,
// one barrier per k-step.
// =====================================================================
template <int CMODE>
__global__ __launch_bounds__(256) void moe_gemm(
    const __bf16* __restrict__ A, const float* __restrict__ B,
    void* __restrict__ C,
    const int* __restrict__ a_gather, const int* __restrict__ c_scatter,
    const int* __restrict__ counts, const int* __restrict__ offsets,
    long long bstride, int Mfixed, int K, int N, int lda, int ldb, int ldc)
{
    const int e   = blockIdx.z;
    const int off = offsets ? offsets[e] : 0;
    const int M   = counts ? counts[e] : Mfixed;
    const int m0  = blockIdx.y * BM;
    if (m0 >= M) return;
    const int n0 = blockIdx.x * BN;
    const float* Bp = B + (long long)e * bstride;

    __shared__ __bf16 As[2][BM][LDK];   // [buf][row][k]
    __shared__ __bf16 Bs[2][BN][LDK];   // [buf][col][k]  (B stored transposed)
    __shared__ int rowmap[BM];

    const int tid = threadIdx.x;
    if (tid < BM) {
        int r = m0 + tid;
        rowmap[tid] = (r < M) ? (a_gather ? a_gather[off + r] : (off + r)) : -1;
    }
    __syncthreads();

    const int wave = tid >> 5, lane = tid & 31;
    const int wm = wave & 1, wn = wave >> 1;        // 2 x 4 wave grid
    const int half = lane >> 4, l16 = lane & 15;

    // global->LDS assignments
    const int ar = tid >> 1;                // A row 0..127
    const int ak = (tid & 1) * 16;          // k offset 0 / 16
    const int bk = tid & 31;                // B k-row
    const int bn = (tid >> 5) * 32;         // B 32-col group

    const int g = rowmap[ar];
    const __bf16* aptr = (g >= 0) ? (A + (long long)g * lda + ak) : nullptr;
    if (!aptr) {  // zero dead rows once, in both buffers
        uint4 z = make_uint4(0u, 0u, 0u, 0u);
        *(uint4*)&As[0][ar][ak] = z; *(uint4*)&As[0][ar][ak + 8] = z;
        *(uint4*)&As[1][ar][ak] = z; *(uint4*)&As[1][ar][ak + 8] = z;
    }

    auto load_tile = [&](int k0, int buf) {
        if (aptr) {
            uint4 d0 = *(const uint4*)(aptr + k0);
            uint4 d1 = *(const uint4*)(aptr + k0 + 8);
            *(uint4*)&As[buf][ar][ak]     = d0;
            *(uint4*)&As[buf][ar][ak + 8] = d1;
        }
        const float* src = Bp + (long long)(k0 + bk) * ldb + n0 + bn;
#pragma unroll
        for (int i = 0; i < 32; i += 4) {
            float4 f = *(const float4*)(src + i);
            Bs[buf][bn + i + 0][bk] = (__bf16)f.x;
            Bs[buf][bn + i + 1][bk] = (__bf16)f.y;
            Bs[buf][bn + i + 2][bk] = (__bf16)f.z;
            Bs[buf][bn + i + 3][bk] = (__bf16)f.w;
        }
    };

    v8f acc[4][4];
#pragma unroll
    for (int mi = 0; mi < 4; ++mi)
#pragma unroll
        for (int ni = 0; ni < 4; ++ni)
#pragma unroll
            for (int j = 0; j < 8; ++j) acc[mi][ni][j] = 0.f;

    // prologue: stage tile 0
    load_tile(0, 0);
    __syncthreads();

    int cur = 0;
    for (int k0 = 0; k0 < K; k0 += BK) {
        if (k0 + BK < K) load_tile(k0 + BK, cur ^ 1);   // overlap with compute

        // ---- fragments (ISA 16-bit A 16x32 / B 32x16 layouts) ----
        v16bf afrag[4];
#pragma unroll
        for (int mi = 0; mi < 4; ++mi) {
            const __bf16* p = &As[cur][wm * 64 + mi * 16 + l16][half * 8];
            v8bf lo = *(const v8bf*)p;          // K 0-7  (half0) / 8-15  (half1)
            v8bf hi = *(const v8bf*)(p + 16);   // K 16-23(half0) / 24-31 (half1)
#pragma unroll
            for (int j = 0; j < 8; ++j) { afrag[mi][j] = lo[j]; afrag[mi][j + 8] = hi[j]; }
        }
#pragma unroll
        for (int ni = 0; ni < 4; ++ni) {
            const __bf16* p = &Bs[cur][wn * 64 + ni * 16 + l16][half * 8];
            v8bf lo = *(const v8bf*)p;          // K half*8 + 0..7
            v8bf hi = *(const v8bf*)(p + 16);   // K 16 + half*8 + 0..7
            v16bf bfrag;
#pragma unroll
            for (int j = 0; j < 8; ++j) { bfrag[j] = lo[j]; bfrag[j + 8] = hi[j]; }
#pragma unroll
            for (int mi = 0; mi < 4; ++mi)
                acc[mi][ni] = __builtin_amdgcn_wmma_f32_16x16x32_bf16(
                    false, afrag[mi], false, bfrag,
                    (short)0, acc[mi][ni], false, false);
        }
        __syncthreads();
        cur ^= 1;
    }

    // ---- epilogue: C 16x16 f32 layout: VGPR v -> M = half*8 + v, N = l16 ----
#pragma unroll
    for (int mi = 0; mi < 4; ++mi) {
#pragma unroll
        for (int v = 0; v < 8; ++v) {
            int lrow = m0 + wm * 64 + mi * 16 + half * 8 + v;
            if (lrow >= M) continue;
            long long crow = c_scatter ? (long long)c_scatter[off + lrow]
                                       : (long long)(off + lrow);
#pragma unroll
            for (int ni = 0; ni < 4; ++ni) {
                int col = n0 + wn * 64 + ni * 16 + l16;
                long long idx = crow * ldc + col;
                if (CMODE == 0)      ((__bf16*)C)[idx]  = (__bf16)acc[mi][ni][v];
                else if (CMODE == 1) ((float*)C)[idx]   = acc[mi][ni][v];
                else                 ((float*)C)[idx]  += acc[mi][ni][v];
            }
        }
    }
}

// =====================================================================
extern "C" void kernel_launch(void* const* d_in, const int* in_sizes, int n_in,
                              void* d_out, int out_size, void* d_ws, size_t ws_size,
                              hipStream_t stream)
{
    const float* x      = (const float*)d_in[0];   // (B,S,D)
    const float* rw     = (const float*)d_in[1];   // (E,D)
    const float* wgu    = (const float*)d_in[2];   // (E,D,2F)
    const float* wdn    = (const float*)d_in[3];   // (E,F,D)
    const float* sgu    = (const float*)d_in[4];   // (D,2F)
    const float* sdn    = (const float*)d_in[5];   // (F,D)
    float* out          = (float*)d_out;           // (B,S,D)

    // ---- workspace layout ----
    char* ws = (char*)d_ws;
    int*   counts  = (int*)ws;                       // 8
    int*   offsets = counts + 8;                     // 8
    int*   cursor  = offsets + 8;                    // 8
    int*   eid     = (int*)(ws + 256);               // T
    float* gate    = (float*)(ws + 256 + (size_t)Tc * 4);
    int*   tlist   = (int*)(ws + 256 + (size_t)Tc * 8);
    __bf16* xb = (__bf16*)(ws + 256 + (size_t)Tc * 12);
    __bf16* xs = xb + (size_t)Tc * Dc;
    __bf16* Hr = xs + (size_t)Tc * Dc;               // T x 2F
    __bf16* Hs = Hr + (size_t)Tc * F2c;              // T x F

    hipMemsetAsync(counts, 0, 8 * sizeof(int), stream);

    router_kernel<<<Tc, 256, 0, stream>>>(x, rw, eid, gate, counts);
    scan_kernel<<<1, 32, 0, stream>>>(counts, offsets, cursor);
    scatter_kernel<<<(Tc + 255) / 256, 256, 0, stream>>>(eid, offsets, cursor, tlist);
    prep_kernel<<<(int)(((long long)Tc * Dc / 4 + 255) / 256), 256, 0, stream>>>(x, gate, xb, xs);

    // ---- routed: Hr = gather(xs) @ wgu[e] ----
    dim3 g1(F2c / BN, Tc / BM, Ec);
    moe_gemm<0><<<g1, 256, 0, stream>>>(xs, wgu, Hr, tlist, nullptr, counts, offsets,
                                        (long long)Dc * F2c, 0, Dc, F2c, Dc, F2c, F2c);
    silu_kernel<<<(int)(((long long)Tc * Fc / 4 + 255) / 256), 256, 0, stream>>>(Hr, Hs);
    // ---- routed: out[tok] = Hs @ wdn[e]  (scatter rows; K=1 => exclusive) ----
    dim3 g2(Dc / BN, Tc / BM, Ec);
    moe_gemm<1><<<g2, 256, 0, stream>>>(Hs, wdn, out, nullptr, tlist, counts, offsets,
                                        (long long)Fc * Dc, 0, Fc, Dc, Fc, Dc, Dc);

    // ---- shared expert: Hr = xb @ sgu ; out += Hs @ sdn ----
    dim3 g3(F2c / BN, Tc / BM, 1);
    moe_gemm<0><<<g3, 256, 0, stream>>>(xb, sgu, Hr, nullptr, nullptr, nullptr, nullptr,
                                        0, Tc, Dc, F2c, Dc, F2c, F2c);
    silu_kernel<<<(int)(((long long)Tc * Fc / 4 + 255) / 256), 256, 0, stream>>>(Hr, Hs);
    dim3 g4(Dc / BN, Tc / BM, 1);
    moe_gemm<2><<<g4, 256, 0, stream>>>(Hs, sdn, out, nullptr, nullptr, nullptr, nullptr,
                                        0, Tc, Fc, Dc, Fc, Dc, Dc);
}
